// LocalCrossAttention_9337258902219
// MI455X (gfx1250) — matrix operand order repaired and checked by reading, not compiled, not run on hardware
//
#include <hip/hip_runtime.h>
#include <hip/hip_bf16.h>

// LocalCrossAttention for MI455X (gfx1250, wave32, WMMA).
//
// Roofline: weights ~201MB -> 8.6us @ 23.3 TB/s HBM; weights fit 192MB L2 so
// M-tile re-reads are L2 hits. Projections (19.3 GFLOP) run as bf16 hi/lo
// split (3x v_wmma_f32_16x16x32_bf16 per fp32 product, ~fp32 accuracy).
// Attention uses exact v_wmma_f32_16x16x4_f32.

typedef __attribute__((ext_vector_type(16))) __bf16 v16bf;
typedef __attribute__((ext_vector_type(8)))  float  v8f;
typedef __attribute__((ext_vector_type(2)))  float  v2f;

union FragB  { uint4 q[2]; v16bf v; };            // 16 bf16 = 32B
union Pack8  { unsigned short u[8]; uint4 q; };   // 8 bf16  = 16B

__device__ __forceinline__ unsigned short f2bf(float x) {
  unsigned u = __float_as_uint(x);
  u += 0x7FFFu + ((u >> 16) & 1u);                // round-to-nearest-even
  return (unsigned short)(u >> 16);
}
__device__ __forceinline__ float bf2f(unsigned short h) {
  return __uint_as_float((unsigned)h << 16);
}

// ---------------------------------------------------------------------------
// Projection GEMM: C[m,n] = sum_f X[m,f] * W[o,n,f] + bias[o,n]
//   mode 0 (Q):  m = b (64 rows),  out idx = (b*4 + o)*2048 + n
//   mode 1 (KV): m = b*4+i (256),  out idx = ((b*4+o)*4 + i)*2048 + n
// Block tile 64x64, BK=32, 256 threads (8 waves); wave w -> rows (w>>1)*16,
// cols (w&1)*32 (two 16x16 accumulators). 3 WMMAs per tile per K-chunk.
// ---------------------------------------------------------------------------
#define GBM 64
#define GBN 64
#define GBK 32

__global__ __launch_bounds__(256)
void proj_gemm_kernel(const float* __restrict__ X, const float* __restrict__ W,
                      const float* __restrict__ bias, float* __restrict__ out,
                      int x_head_stride, int x_row_stride, int mode)
{
  const int o  = blockIdx.z;
  const int n0 = blockIdx.x * GBN;
  const int m0 = blockIdx.y * GBM;
  const int Kdim = 2048, Ndim = 2048;

  __shared__ __align__(16) unsigned short Ahi[GBM][GBK];
  __shared__ __align__(16) unsigned short Alo[GBM][GBK];
  __shared__ __align__(16) unsigned short Bhi[GBN][GBK];
  __shared__ __align__(16) unsigned short Blo[GBN][GBK];

  const int t  = threadIdx.x;
  const int sr = t >> 2;                 // staging row 0..63
  const int sc = (t & 3) << 3;           // staging col {0,8,16,24}

  const float* xrow = X + (size_t)o * x_head_stride
                        + (size_t)(m0 + sr) * x_row_stride + sc;
  const float* wrow = W + ((size_t)o * Ndim + (n0 + sr)) * (size_t)Kdim + sc;

  const int wave  = t >> 5;
  const int lane  = t & 31;
  const int mbase = (wave >> 1) << 4;    // 0,16,32,48
  const int nbase = (wave & 1) << 5;     // 0,32

  v8f acc0 = {0.f,0.f,0.f,0.f,0.f,0.f,0.f,0.f};
  v8f acc1 = acc0;

  float4 ax0, ax1, bx0, bx1;
  ax0 = *(const float4*)(xrow);
  ax1 = *(const float4*)(xrow + 4);
  bx0 = *(const float4*)(wrow);
  bx1 = *(const float4*)(wrow + 4);

  const int nk = Kdim / GBK;             // 64 chunks
  for (int kc = 0; kc < nk; ++kc) {
    // --- convert prefetched f32 to bf16 hi/lo and stage into LDS ---
    {
      float a8[8] = {ax0.x,ax0.y,ax0.z,ax0.w, ax1.x,ax1.y,ax1.z,ax1.w};
      float b8[8] = {bx0.x,bx0.y,bx0.z,bx0.w, bx1.x,bx1.y,bx1.z,bx1.w};
      Pack8 ah, al, bh, bl;
      #pragma unroll
      for (int e = 0; e < 8; ++e) {
        unsigned short h = f2bf(a8[e]);
        ah.u[e] = h; al.u[e] = f2bf(a8[e] - bf2f(h));
        unsigned short g = f2bf(b8[e]);
        bh.u[e] = g; bl.u[e] = f2bf(b8[e] - bf2f(g));
      }
      *(uint4*)&Ahi[sr][sc] = ah.q;  *(uint4*)&Alo[sr][sc] = al.q;
      *(uint4*)&Bhi[sr][sc] = bh.q;  *(uint4*)&Blo[sr][sc] = bl.q;
    }
    __syncthreads();

    // --- prefetch next chunk while WMMAs run ---
    if (kc + 1 < nk) {
      const float* xn = xrow + (size_t)(kc + 1) * GBK;
      const float* wn = wrow + (size_t)(kc + 1) * GBK;
      ax0 = *(const float4*)(xn);      ax1 = *(const float4*)(xn + 4);
      bx0 = *(const float4*)(wn);      bx1 = *(const float4*)(wn + 4);
    }

    // --- fragments (16-bit A 16x32 / B 32x16, contiguous k = 16h + s in both) ---
    {
      const int m  = mbase + (lane & 15);
      const int kh = (lane >> 4) << 4;           // A: K base 0 / 16
      FragB a_hi, a_lo;
      a_hi.q[0] = *(const uint4*)&Ahi[m][kh];
      a_hi.q[1] = *(const uint4*)&Ahi[m][kh + 8];
      a_lo.q[0] = *(const uint4*)&Alo[m][kh];
      a_lo.q[1] = *(const uint4*)&Alo[m][kh + 8];

      const int n  = nbase + (lane & 15);
      const int kb = (lane >> 4) << 4;           // B: K base 0 / 16
      FragB b_hi0, b_lo0, b_hi1, b_lo1;
      b_hi0.q[0] = *(const uint4*)&Bhi[n][kb];
      b_hi0.q[1] = *(const uint4*)&Bhi[n][kb + 8];
      b_lo0.q[0] = *(const uint4*)&Blo[n][kb];
      b_lo0.q[1] = *(const uint4*)&Blo[n][kb + 8];
      b_hi1.q[0] = *(const uint4*)&Bhi[n + 16][kb];
      b_hi1.q[1] = *(const uint4*)&Bhi[n + 16][kb + 8];
      b_lo1.q[0] = *(const uint4*)&Blo[n + 16][kb];
      b_lo1.q[1] = *(const uint4*)&Blo[n + 16][kb + 8];

      // (hi*hi + lo*hi + hi*lo): ~fp32 accuracy with fp32 accumulate
      acc0 = __builtin_amdgcn_wmma_f32_16x16x32_bf16(false, a_hi.v, false, b_hi0.v, (short)0, acc0, false, false);
      acc0 = __builtin_amdgcn_wmma_f32_16x16x32_bf16(false, a_lo.v, false, b_hi0.v, (short)0, acc0, false, false);
      acc0 = __builtin_amdgcn_wmma_f32_16x16x32_bf16(false, a_hi.v, false, b_lo0.v, (short)0, acc0, false, false);
      acc1 = __builtin_amdgcn_wmma_f32_16x16x32_bf16(false, a_hi.v, false, b_hi1.v, (short)0, acc1, false, false);
      acc1 = __builtin_amdgcn_wmma_f32_16x16x32_bf16(false, a_lo.v, false, b_hi1.v, (short)0, acc1, false, false);
      acc1 = __builtin_amdgcn_wmma_f32_16x16x32_bf16(false, a_hi.v, false, b_lo1.v, (short)0, acc1, false, false);
    }
    __syncthreads();
  }

  // --- epilogue: C/D layout: lane holds col n=lane&15, rows (lane>>4)*8 + r
  #pragma unroll
  for (int tt = 0; tt < 2; ++tt) {
    v8f acc = tt ? acc1 : acc0;
    const int gn = n0 + nbase + tt * 16 + (lane & 15);
    const float bv = bias[(size_t)o * 2048 + gn];
    #pragma unroll
    for (int r = 0; r < 8; ++r) {
      const int gm = m0 + mbase + ((lane >> 4) << 3) + r;
      size_t idx;
      if (mode == 0) {                        // Q: [b][o][f]
        idx = ((size_t)gm * 4 + o) * 2048 + gn;
      } else {                                // KV: rows are b*4+i -> [b][o][i][f]
        const int b = gm >> 2, i = gm & 3;
        idx = (((size_t)b * 4 + o) * 4 + (size_t)i) * 2048 + gn;
      }
      out[idx] = acc[r] + bv;
    }
  }
}

// ---------------------------------------------------------------------------
// Attention: one block per (b,o), 128 threads (4 waves).
// scores = Q @ K^T (exact f32 WMMA 16x16x4), SafeSoftmax e/(sum(e)+1e-5),
// Z += attn @ V accumulated over i, out = Z / sqrt(64).
// ---------------------------------------------------------------------------
__global__ __launch_bounds__(128)
void attn_kernel(const float* __restrict__ Qw, const float* __restrict__ Kw,
                 const float* __restrict__ Vw, float* __restrict__ out)
{
  const int bo = blockIdx.x;                 // b*4 + o
  __shared__ __align__(16) float Qs[32][64];
  __shared__ __align__(16) float Ks[32][64];
  __shared__ __align__(16) float Vs[32][64];
  __shared__ float Es[32][32];
  __shared__ float Rs[32];

  const int t = threadIdx.x;
  const int wave = t >> 5, lane = t & 31;
  const int mw = wave >> 1, nw = wave & 1;

  // stage Q tile (2048 f32)
  {
    const float4* src = (const float4*)(Qw + (size_t)bo * 2048);
    float4* dst = (float4*)&Qs[0][0];
    #pragma unroll
    for (int j = 0; j < 4; ++j) dst[t + 128 * j] = src[t + 128 * j];
  }

  v8f z0 = {0.f,0.f,0.f,0.f,0.f,0.f,0.f,0.f};
  v8f z1 = z0;

  for (int i = 0; i < 4; ++i) {
    __syncthreads();                         // protect Ks/Vs re-staging
    {
      const float4* ks = (const float4*)(Kw + ((size_t)bo * 4 + i) * 2048);
      const float4* vs = (const float4*)(Vw + ((size_t)bo * 4 + i) * 2048);
      float4* kd = (float4*)&Ks[0][0];
      float4* vd = (float4*)&Vs[0][0];
      #pragma unroll
      for (int j = 0; j < 4; ++j) {
        kd[t + 128 * j] = ks[t + 128 * j];
        vd[t + 128 * j] = vs[t + 128 * j];
      }
    }
    __syncthreads();

    // scores tile (mw,nw), contraction over d=64 in K=4 steps
    {
      v8f s = {0.f,0.f,0.f,0.f,0.f,0.f,0.f,0.f};
      const int m  = (mw << 4) + (lane & 15);
      const int n  = (nw << 4) + (lane & 15);
      const int e2 = (lane >> 4) << 1;       // K element pair 0 / 2
      #pragma unroll
      for (int d0 = 0; d0 < 64; d0 += 4) {
        v2f a  = *(const v2f*)&Qs[m][d0 + e2];
        v2f bb = *(const v2f*)&Ks[n][d0 + e2];
        s = __builtin_amdgcn_wmma_f32_16x16x4_f32(false, a, false, bb, (short)0, s, false, false);
      }
      #pragma unroll
      for (int r = 0; r < 8; ++r)
        Es[(mw << 4) + ((lane >> 4) << 3) + r][(nw << 4) + (lane & 15)] = expf(s[r]);
    }
    __syncthreads();

    if (t < 32) {                            // row sums over 32 cols
      float sum = 0.f;
      #pragma unroll
      for (int c = 0; c < 32; ++c) sum += Es[t][c];
      Rs[t] = sum;
    }
    __syncthreads();

    // Z += attn @ V ; normalization folded into A-fragment (fixed row per lane)
    {
      const int m   = (mw << 4) + (lane & 15);
      const float rcp = 1.0f / (Rs[m] + 1e-5f);
      const int n0c = ((wave & 1) << 5) + (lane & 15);
      #pragma unroll
      for (int k0 = 0; k0 < 32; k0 += 4) {
        const int ke = k0 + ((lane >> 4) << 1);
        v2f a;  a.x  = Es[m][ke] * rcp;  a.y  = Es[m][ke + 1] * rcp;
        v2f b0; b0.x = Vs[ke][n0c];      b0.y = Vs[ke + 1][n0c];
        v2f b1; b1.x = Vs[ke][n0c + 16]; b1.y = Vs[ke + 1][n0c + 16];
        z0 = __builtin_amdgcn_wmma_f32_16x16x4_f32(false, a, false, b0, (short)0, z0, false, false);
        z1 = __builtin_amdgcn_wmma_f32_16x16x4_f32(false, a, false, b1, (short)0, z1, false, false);
      }
    }
  }

  // epilogue: out[(b*4+o)*32 + q][d], coeff = 1/sqrt(64)
  {
    const int q0 = (mw << 4) + ((lane >> 4) << 3);
    const int d0 = ((wave & 1) << 5) + (lane & 15);
    float* ob = out + (size_t)bo * 32 * 64;
    #pragma unroll
    for (int r = 0; r < 8; ++r) {
      ob[(size_t)(q0 + r) * 64 + d0]      = 0.125f * z0[r];
      ob[(size_t)(q0 + r) * 64 + d0 + 16] = 0.125f * z1[r];
    }
  }
}

// ---------------------------------------------------------------------------
// Launch. Workspace: Q 2MB | K 8MB | V 8MB (18MB total, f32).
// ---------------------------------------------------------------------------
extern "C" void kernel_launch(void* const* d_in, const int* in_sizes, int n_in,
                              void* d_out, int out_size, void* d_ws, size_t ws_size,
                              hipStream_t stream) {
  const float* dec = (const float*)d_in[0];   // [64,128,64]
  const float* enc = (const float*)d_in[1];   // [64,128,64]
  const float* WQw = (const float*)d_in[2];   // [4,2048,2048]
  const float* WQb = (const float*)d_in[3];   // [4,2048]
  const float* WKw = (const float*)d_in[4];
  const float* WKb = (const float*)d_in[5];
  const float* WVw = (const float*)d_in[6];
  const float* WVb = (const float*)d_in[7];
  float* out = (float*)d_out;

  float* Qws = (float*)d_ws;                  //  524288 f32
  float* Kws = Qws + 524288;                  // 2097152 f32
  float* Vws = Kws + 2097152;                 // 2097152 f32

  dim3 blk(256);
  // Q: M=64 rows (b), x offset per head = 2048, row stride = 8192
  proj_gemm_kernel<<<dim3(32, 1, 4), blk, 0, stream>>>(dec, WQw, WQb, Qws, 2048, 8192, 0);
  // K/V: M=256 rows (b*4+i), row stride = 2048
  proj_gemm_kernel<<<dim3(32, 4, 4), blk, 0, stream>>>(enc, WKw, WKb, Kws, 0, 2048, 1);
  proj_gemm_kernel<<<dim3(32, 4, 4), blk, 0, stream>>>(enc, WVw, WVb, Vws, 0, 2048, 1);

  attn_kernel<<<dim3(256), dim3(128), 0, stream>>>(Qws, Kws, Vws, out);
}
